// MultiDimWindowAttention_1494648619770
// MI455X (gfx1250) — matrix operand order repaired and checked by reading, not compiled, not run on hardware
//
#include <hip/hip_runtime.h>

// ---------------------------------------------------------------------------
// MultiDimWindowAttention for MI455X (gfx1250, wave32, WMMA 16x16x32 f16)
//
// Pipeline (all f16 storage for intermediates, f32 WMMA accumulation):
//   1. wt_f16_kernel      : W (KxN f32) -> W^T (NxK f16)  (x3 weights)
//   2. qkv_gemm_kernel    : x @ Wqkv + b -> qw/qr (scaled q + rel biases), k, v (f16)
//   3. rp_gemm_kernel     : r @ Wemb + b -> rp (f16)
//   4. attn_kernel        : per (s, head): WMMA scores (AC+BD packed in one
//                           16x192 pass), joint softmax over win*C=168,
//                           WMMA prob @ V -> z (f16)
//   5. out_gemm_kernel    : z @ Wout + b -> out (f32)
// ---------------------------------------------------------------------------

#define LSEQ   1024
#define CCELL  8
#define NHEAD  8
#define HDIM   32
#define DMODEL 256   // NHEAD*HDIM
#define H1DIM  256
#define H2DIM  256
#define WINSZ  21
#define WHALF  10
#define KEYC   168   // WINSZ*CCELL
#define NCOLS  192   // 168 key cols + 21 rp cols + 3 pad (12 tiles of 16)

typedef _Float16 f16;
typedef _Float16 v16h __attribute__((ext_vector_type(16)));
typedef _Float16 v8h  __attribute__((ext_vector_type(8)));
typedef float    v8f  __attribute__((ext_vector_type(8)));

static __device__ inline v16h pack16(v8h lo, v8h hi) {
  v16h o;
#pragma unroll
  for (int i = 0; i < 8; ++i) { o[i] = lo[i]; o[i + 8] = hi[i]; }
  return o;
}

static __device__ inline v8f zero8f() {
  v8f c;
#pragma unroll
  for (int i = 0; i < 8; ++i) c[i] = 0.0f;
  return c;
}

static __device__ inline v16h zero16h() {
  v16h c;
#pragma unroll
  for (int i = 0; i < 16; ++i) c[i] = (f16)0.0f;
  return c;
}

// ---------------------------------------------------------------------------
// Weight transpose + f32 -> f16 convert:  Wt[n*K + k] = (f16) W[k*N + n]
// ---------------------------------------------------------------------------
__global__ void wt_f16_kernel(const float* __restrict__ W, f16* __restrict__ Wt,
                              int K, int N) {
  int idx = blockIdx.x * blockDim.x + threadIdx.x;
  if (idx >= K * N) return;
  int n = idx / K;
  int k = idx - n * K;
  Wt[idx] = (f16)W[(size_t)k * N + n];
}

// ---------------------------------------------------------------------------
// QKV GEMM: X[8192x256] f32  x  WqkvT[768x256] f16  (+ b_qkv)
// Epilogue splits into:
//   qw = (q+b)*scale + bias_r_w   qr = (q+b)*scale + bias_r_r   (f16)
//   kh = k+b, vh = v+b                                          (f16)
// One wave per 16x16 tile, K-loop of 8 WMMA steps.
// ---------------------------------------------------------------------------
__global__ __launch_bounds__(128)
void qkv_gemm_kernel(const float* __restrict__ X, const f16* __restrict__ Bt,
                     const float* __restrict__ bqkv,
                     const float* __restrict__ brw, const float* __restrict__ brr,
                     f16* __restrict__ qw, f16* __restrict__ qr,
                     f16* __restrict__ kh, f16* __restrict__ vh)
{
  const int K = H1DIM, NT = 48;                 // 768/16 column tiles
  const int lane = threadIdx.x & 31;
  const int wid  = blockIdx.x * 4 + (threadIdx.x >> 5);
  const int mt = wid / NT, nt = wid - mt * NT;
  const int r = lane & 15, hl = lane >> 4;
  const int kb = hl * 8, kb16 = hl * 16;

  const float* arow = X  + (size_t)(mt * 16 + r) * K;
  const f16*   brow = Bt + (size_t)(nt * 16 + r) * K;

  v8f acc = zero8f();
#pragma unroll
  for (int kk = 0; kk < K; kk += 32) {
    v16h a;
#pragma unroll
    for (int i = 0; i < 8; ++i) {               // A: f32 -> f16 on the fly
      a[i]     = (f16)arow[kk + kb + i];
      a[i + 8] = (f16)arow[kk + 16 + kb + i];
    }
    v8h b0 = *(const v8h*)(brow + kk + kb);
    v8h b1 = *(const v8h*)(brow + kk + 16 + kb);
    v16h b = pack16(b0, b1);
    acc = __builtin_amdgcn_wmma_f32_16x16x32_f16(false, a, false, b,
                                                 (short)0, acc, false, false);
  }

  const int col = nt * 16 + r;                  // 0..767
  const int region = col >> 8;                  // 0=q 1=k 2=v
  const int inner  = col & 255;
  const float bq = bqkv[col];
  const int rowbase = mt * 16 + hl * 8;
  const float scale = 0.17677669529663689f;     // 1/sqrt(HD)

  if (region == 0) {
    const float bw = brw[inner], br = brr[inner];
#pragma unroll
    for (int g = 0; g < 8; ++g) {
      float v = (acc[g] + bq) * scale;
      size_t o = (size_t)(rowbase + g) * DMODEL + inner;
      qw[o] = (f16)(v + bw);
      qr[o] = (f16)(v + br);
    }
  } else if (region == 1) {
#pragma unroll
    for (int g = 0; g < 8; ++g)
      kh[(size_t)(rowbase + g) * DMODEL + inner] = (f16)(acc[g] + bq);
  } else {
#pragma unroll
    for (int g = 0; g < 8; ++g)
      vh[(size_t)(rowbase + g) * DMODEL + inner] = (f16)(acc[g] + bq);
  }
}

// ---------------------------------------------------------------------------
// Rel-emb GEMM: R[21504x256] f32  x  WembT[256x256] f16  (+ b_emb) -> rp f16
// ---------------------------------------------------------------------------
__global__ __launch_bounds__(128)
void rp_gemm_kernel(const float* __restrict__ R, const f16* __restrict__ Bt,
                    const float* __restrict__ bemb, f16* __restrict__ rp)
{
  const int K = H1DIM, NT = 16;
  const int lane = threadIdx.x & 31;
  const int wid  = blockIdx.x * 4 + (threadIdx.x >> 5);
  const int mt = wid / NT, nt = wid - mt * NT;
  const int r = lane & 15, hl = lane >> 4;
  const int kb = hl * 8, kb16 = hl * 16;

  const float* arow = R  + (size_t)(mt * 16 + r) * K;
  const f16*   brow = Bt + (size_t)(nt * 16 + r) * K;

  v8f acc = zero8f();
#pragma unroll
  for (int kk = 0; kk < K; kk += 32) {
    v16h a;
#pragma unroll
    for (int i = 0; i < 8; ++i) {
      a[i]     = (f16)arow[kk + kb + i];
      a[i + 8] = (f16)arow[kk + 16 + kb + i];
    }
    v8h b0 = *(const v8h*)(brow + kk + kb);
    v8h b1 = *(const v8h*)(brow + kk + 16 + kb);
    v16h b = pack16(b0, b1);
    acc = __builtin_amdgcn_wmma_f32_16x16x32_f16(false, a, false, b,
                                                 (short)0, acc, false, false);
  }

  const int col = nt * 16 + r;
  const float bb = bemb[col];
  const int rowbase = mt * 16 + hl * 8;
#pragma unroll
  for (int g = 0; g < 8; ++g)
    rp[(size_t)(rowbase + g) * DMODEL + col] = (f16)(acc[g] + bb);
}

// ---------------------------------------------------------------------------
// Attention core: one wave per (s, head).
//  A-matrix (16x32): rows 0..7 = qw (q*scale+bias_r_w), rows 8..15 = qr.
//  B-matrix cols: 0..167 windowed keys (j,d), 168..188 rp vectors, rest 0.
//  -> rows 0..7 of key-cols = AC term, rows 8..15 of rp-cols = BD term.
//  Joint softmax over 168 (torch flatten order j*C+d), P f16 zero-padded to
//  K=192, then z = P @ V with V staged transposed in LDS.
// ---------------------------------------------------------------------------
__global__ __launch_bounds__(32)
void attn_kernel(const f16* __restrict__ qw, const f16* __restrict__ qr,
                 const f16* __restrict__ kh, const f16* __restrict__ vh,
                 const f16* __restrict__ rp, f16* __restrict__ zh)
{
  __shared__ __attribute__((aligned(16))) float S[16][NCOLS];   // scores
  __shared__ __attribute__((aligned(16))) f16   P[16][NCOLS];   // probs
  __shared__ __attribute__((aligned(16))) f16   Vt[HDIM][NCOLS]; // V^T

  const int s = blockIdx.x;
  const int n = blockIdx.y;
  const int lane = threadIdx.x;
  const int r = lane & 15, hl = lane >> 4;
  const int kb = hl * 8, kb16 = hl * 16;

  // zero-init probability matrix (rows 8..15 and cols >=168 stay zero)
  {
    unsigned int* p32 = (unsigned int*)&P[0][0];
#pragma unroll
    for (int i = 0; i < (16 * NCOLS * 2 / 4) / 32; ++i)
      p32[lane + i * 32] = 0u;
  }

  // stage windowed V transposed: Vt[h][jd] = v[s+j-w, d, n, h], 0 when OOB
  for (int jd = lane; jd < NCOLS; jd += 32) {
    int j = jd >> 3, d = jd & 7;
    int pos = s + j - WHALF;
    if (jd < KEYC && pos >= 0 && pos < LSEQ) {
      const f16* src = vh + ((size_t)(pos * CCELL + d) * DMODEL + n * HDIM);
      v8h v0 = *(const v8h*)(src);
      v8h v1 = *(const v8h*)(src + 8);
      v8h v2 = *(const v8h*)(src + 16);
      v8h v3 = *(const v8h*)(src + 24);
#pragma unroll
      for (int h = 0; h < 8; ++h) {
        Vt[h][jd]      = v0[h];
        Vt[h + 8][jd]  = v1[h];
        Vt[h + 16][jd] = v2[h];
        Vt[h + 24][jd] = v3[h];
      }
    } else {
#pragma unroll
      for (int h = 0; h < HDIM; ++h) Vt[h][jd] = (f16)0.0f;
    }
  }

  // A fragment: row r<8 from qw, rows 8..15 from qr
  v16h aq;
  {
    const f16* src = (r < 8)
        ? (qw + ((size_t)(s * CCELL + r) * DMODEL + n * HDIM))
        : (qr + ((size_t)(s * CCELL + (r - 8)) * DMODEL + n * HDIM));
    v8h a0 = *(const v8h*)(src + kb16);
    v8h a1 = *(const v8h*)(src + kb16 + 8);
    aq = pack16(a0, a1);
  }

  // scores: 12 column tiles of 16, single K-step (K = HD = 32)
  for (int t = 0; t < 12; ++t) {
    const int col = t * 16 + r;
    const f16* src = nullptr;
    if (col < KEYC) {
      int j = col >> 3, d = col & 7;
      int pos = s + j - WHALF;
      if (pos >= 0 && pos < LSEQ)
        src = kh + ((size_t)(pos * CCELL + d) * DMODEL + n * HDIM + kb16);
    } else if (col < KEYC + WINSZ) {
      int j = col - KEYC;
      src = rp + ((size_t)(j * LSEQ + s) * DMODEL + n * HDIM + kb16);
    }
    v16h b;
    if (src) {
      v8h b0 = *(const v8h*)(src);
      v8h b1 = *(const v8h*)(src + 8);
      b = pack16(b0, b1);
    } else {
      b = zero16h();
    }
    v8f c = zero8f();
    c = __builtin_amdgcn_wmma_f32_16x16x32_f16(false, aq, false, b,
                                               (short)0, c, false, false);
#pragma unroll
    for (int g = 0; g < 8; ++g) S[hl * 8 + g][col] = c[g];
  }
  __syncthreads();

  // joint softmax over win*C = 168 entries per row c (4 lanes per row)
  {
    const int row = lane >> 2, sub = lane & 3;
    float mx = -3.402823466e38f;
    for (int col = sub; col < KEYC; col += 4) {
      int j = col >> 3;
      int pos = s + j - WHALF;
      float v = (pos >= 0 && pos < LSEQ)
                    ? (S[row][col] + S[8 + row][KEYC + j])   // AC + BD
                    : -3.402823466e38f;
      S[row][col] = v;
      mx = fmaxf(mx, v);
    }
    mx = fmaxf(mx, __shfl_xor(mx, 1, 4));
    mx = fmaxf(mx, __shfl_xor(mx, 2, 4));
    float sum = 0.0f;
    for (int col = sub; col < KEYC; col += 4) {
      float v = S[row][col];
      float e = (v > -1.0e37f) ? __expf(v - mx) : 0.0f;
      S[row][col] = e;
      sum += e;
    }
    sum += __shfl_xor(sum, 1, 4);
    sum += __shfl_xor(sum, 2, 4);
    float inv = 1.0f / sum;
    for (int col = sub; col < KEYC; col += 4)
      P[row][col] = (f16)(S[row][col] * inv);
  }
  __syncthreads();

  // z = P(16x192) @ V(192x32): 2 output tiles, 6 WMMA K-steps each
  for (int ntile = 0; ntile < 2; ++ntile) {
    v8f acc = zero8f();
#pragma unroll
    for (int kk = 0; kk < NCOLS; kk += 32) {
      const f16* ap = &P[r][kk + kb];
      v8h a0 = *(const v8h*)(ap);
      v8h a1 = *(const v8h*)(ap + 16);
      v16h a = pack16(a0, a1);
      const f16* bp = &Vt[ntile * 16 + r][kk + kb];
      v8h b0 = *(const v8h*)(bp);
      v8h b1 = *(const v8h*)(bp + 16);
      v16h b = pack16(b0, b1);
      acc = __builtin_amdgcn_wmma_f32_16x16x32_f16(false, a, false, b,
                                                   (short)0, acc, false, false);
    }
    if (hl == 0) {                               // rows 0..7 hold the result
#pragma unroll
      for (int g = 0; g < 8; ++g)
        zh[(size_t)(s * CCELL + g) * DMODEL + n * HDIM + ntile * 16 + r] =
            (f16)acc[g];
    }
  }
}

// ---------------------------------------------------------------------------
// Output GEMM: Z[8192x256] f16  x  WoutT[256x256] f16  (+ b_out) -> out f32
// ---------------------------------------------------------------------------
__global__ __launch_bounds__(128)
void out_gemm_kernel(const f16* __restrict__ Z, const f16* __restrict__ Bt,
                     const float* __restrict__ bout, float* __restrict__ out)
{
  const int K = DMODEL, NT = 16;
  const int lane = threadIdx.x & 31;
  const int wid  = blockIdx.x * 4 + (threadIdx.x >> 5);
  const int mt = wid / NT, nt = wid - mt * NT;
  const int r = lane & 15, hl = lane >> 4;
  const int kb = hl * 8, kb16 = hl * 16;

  const f16* arow = Z  + (size_t)(mt * 16 + r) * K;
  const f16* brow = Bt + (size_t)(nt * 16 + r) * K;

  v8f acc = zero8f();
#pragma unroll
  for (int kk = 0; kk < K; kk += 32) {
    v8h a0 = *(const v8h*)(arow + kk + kb);
    v8h a1 = *(const v8h*)(arow + kk + 16 + kb);
    v16h a = pack16(a0, a1);
    v8h b0 = *(const v8h*)(brow + kk + kb);
    v8h b1 = *(const v8h*)(brow + kk + 16 + kb);
    v16h b = pack16(b0, b1);
    acc = __builtin_amdgcn_wmma_f32_16x16x32_f16(false, a, false, b,
                                                 (short)0, acc, false, false);
  }

  const int col = nt * 16 + r;
  const float bb = bout[col];
  const int rowbase = mt * 16 + hl * 8;
#pragma unroll
  for (int g = 0; g < 8; ++g)
    out[(size_t)(rowbase + g) * H2DIM + col] = acc[g] + bb;
}

// ---------------------------------------------------------------------------
// Host-side launcher
// ---------------------------------------------------------------------------
extern "C" void kernel_launch(void* const* d_in, const int* in_sizes, int n_in,
                              void* d_out, int out_size, void* d_ws, size_t ws_size,
                              hipStream_t stream)
{
  (void)in_sizes; (void)n_in; (void)out_size; (void)ws_size;

  const float* x    = (const float*)d_in[0];   // [1,1024,8,256]
  const float* rr   = (const float*)d_in[1];   // [1,21,1024,256]
  const float* Wqkv = (const float*)d_in[2];   // [256,768]
  const float* bqkv = (const float*)d_in[3];   // [768]
  const float* Wemb = (const float*)d_in[4];   // [256,256]
  const float* bemb = (const float*)d_in[5];   // [256]
  const float* Wout = (const float*)d_in[6];   // [256,256]
  const float* bout = (const float*)d_in[7];   // [256]
  const float* brw  = (const float*)d_in[8];   // [8,32]
  const float* brr  = (const float*)d_in[9];   // [8,32]
  float* out = (float*)d_out;                  // [1,1024,8,256] f32

  // workspace carve-out (~31.2 MB, all chunks 256B-aligned)
  char* ws = (char*)d_ws;
  f16* WqkvT = (f16*)ws; ws += (size_t)768 * 256 * 2;
  f16* WembT = (f16*)ws; ws += (size_t)256 * 256 * 2;
  f16* WoutT = (f16*)ws; ws += (size_t)256 * 256 * 2;
  f16* qw    = (f16*)ws; ws += (size_t)8192 * 256 * 2;
  f16* qr    = (f16*)ws; ws += (size_t)8192 * 256 * 2;
  f16* kh    = (f16*)ws; ws += (size_t)8192 * 256 * 2;
  f16* vh    = (f16*)ws; ws += (size_t)8192 * 256 * 2;
  f16* rp    = (f16*)ws; ws += (size_t)21504 * 256 * 2;
  f16* zh    = (f16*)ws; ws += (size_t)8192 * 256 * 2;

  // 1. weight transposes (f32 -> f16)
  wt_f16_kernel<<<(768 * 256 + 255) / 256, 256, 0, stream>>>(Wqkv, WqkvT, 256, 768);
  wt_f16_kernel<<<(256 * 256 + 255) / 256, 256, 0, stream>>>(Wemb, WembT, 256, 256);
  wt_f16_kernel<<<(256 * 256 + 255) / 256, 256, 0, stream>>>(Wout, WoutT, 256, 256);

  // 2. qkv projection (+ q scaling and rel-bias folding)
  qkv_gemm_kernel<<<(512 * 48) / 4, 128, 0, stream>>>(x, WqkvT, bqkv, brw, brr,
                                                      qw, qr, kh, vh);
  // 3. relative position embedding projection
  rp_gemm_kernel<<<(1344 * 16) / 4, 128, 0, stream>>>(rr, WembT, bemb, rp);

  // 4. windowed attention core (one wave per (s, head))
  attn_kernel<<<dim3(LSEQ, NHEAD), 32, 0, stream>>>(qw, qr, kh, vh, rp, zh);

  // 5. output projection
  out_gemm_kernel<<<(512 * 16) / 4, 128, 0, stream>>>(zh, WoutT, bout, out);
}
